// MyModelWithDynamicWeights_52716428591665
// MI455X (gfx1250) — matrix operand order repaired and checked by reading, not compiled, not run on hardware
//
#include <hip/hip_runtime.h>
#include <hip/hip_bf16.h>
#include <math.h>

// ---------------------------------------------------------------------------
// Problem sizes (compile-time constants from the reference)
// ---------------------------------------------------------------------------
#define B_  32
#define T_  1024
#define I_  64
#define H_  512
#define F_  63
#define G4_ 2048   // 4*H

typedef __bf16 bf16;
typedef __attribute__((ext_vector_type(16))) __bf16 v16bf;
typedef __attribute__((ext_vector_type(8)))  __bf16 v8bf;
typedef __attribute__((ext_vector_type(8)))  float  v8f;
typedef unsigned int v4u __attribute__((ext_vector_type(4)));
typedef int          v4i __attribute__((ext_vector_type(4)));
typedef int          v8i __attribute__((ext_vector_type(8)));

// Tensor Data Mover availability (probe: 5 args on ROCm7.2/clang-22,
// 6 args on amdgpu-toolchain/clang-23).
#if defined(__has_builtin)
#  if __has_builtin(__builtin_amdgcn_tensor_load_to_lds)
#    define HAVE_TDM 1
#  endif
#endif
#ifndef HAVE_TDM
#  define HAVE_TDM 0
#endif

// ---------------------------------------------------------------------------
// WMMA helpers (CDNA5 gfx1250, wave32).
// A tile is 16x32 (MxK) bf16: lane L<16 holds row m0+L, K packs {0..7,16..23};
// lanes 16..31 hold the same rows with K packs {8..15,24..31}.
// B tile is 32x16 (KxN) bf16: lane L holds column n0+(L&15); K packs as for A
// ({0..7,16..23} for L<16, {8..15,24..31} for L>=16).
// C/D: lane L holds column n0+(L&15); VGPR v holds row m0+v (+8 for L>=16).
// ---------------------------------------------------------------------------
__device__ inline v16bf load_a_tile(const bf16* base, int ld, int m0, int k0, int lane) {
  const int row = m0 + (lane & 15);
  const int kb  = (lane < 16) ? 0 : 8;
  const v8bf lo = *(const v8bf*)(base + (long)row * ld + k0 + kb);
  const v8bf hi = *(const v8bf*)(base + (long)row * ld + k0 + 16 + kb);
  v16bf r;
#pragma unroll
  for (int i = 0; i < 8; ++i) { r[i] = lo[i]; r[8 + i] = hi[i]; }
  return r;
}

__device__ inline v16bf load_b_tile(const bf16* base, int ld, int k0, int n0, int lane) {
  const int col = n0 + (lane & 15);
  const int kb  = (lane < 16) ? 0 : 8;
  v16bf r;
#pragma unroll
  for (int i = 0; i < 8; ++i) {
    r[i]     = base[(long)(k0 + kb + i) * ld + col];
    r[8 + i] = base[(long)(k0 + 16 + kb + i) * ld + col];
  }
  return r;
}

__device__ inline v8f wmma_bf16(v16bf a, v16bf b, v8f c) {
  return __builtin_amdgcn_wmma_f32_16x16x32_bf16(false, a, false, b, (short)0, c, false, false);
}

// ---------------------------------------------------------------------------
// TDM: issue one 2D bf16 tile DMA global->LDS.
// D# per CDNA5 ISA:
//  group0: [1:0]=count(1), [63:32]=lds_addr, [120:64]=global_addr, [127:126]=2
//  group1: [17:16]=data_size(1=2B), [79:48]=tensor_dim0, [111:80]=tensor_dim1,
//          [127:112]=tile_dim0, [143:128]=tile_dim1, [207:160]=dim0_stride
// Must be called wave-uniformly (EXEC is ignored by TDM); tracked by TENSORcnt.
// ---------------------------------------------------------------------------
#if HAVE_TDM
__device__ inline void tdm_load_2d_bf16(unsigned lds_byte_addr, const bf16* gsrc,
                                        unsigned cols, unsigned rows,
                                        unsigned row_stride_elems) {
  const unsigned long long ga = (unsigned long long)(uintptr_t)gsrc;
  v4u g0;
  g0[0] = 1u;                                            // count=1, user descriptor
  g0[1] = lds_byte_addr;                                 // LDS dest (bytes)
  g0[2] = (unsigned)ga;                                  // global addr [31:0]
  g0[3] = (unsigned)((ga >> 32) & 0x01FFFFFFu) | (2u << 30); // addr[56:32], type=2
  v8i g1;
  g1[0] = (int)(1u << 16);                               // mask=0, data_size=2B
  g1[1] = (int)((cols & 0xFFFFu) << 16);                 // tensor_dim0[15:0]
  g1[2] = (int)((cols >> 16) & 0xFFFFu) |                // tensor_dim0[31:16]
          (int)((rows & 0xFFFFu) << 16);                 // tensor_dim1[15:0]
  g1[3] = (int)((rows >> 16) & 0xFFFFu) |                // tensor_dim1[31:16]
          (int)(cols << 16);                             // tile_dim0
  g1[4] = (int)(rows & 0xFFFFu);                         // tile_dim1 (tile_dim2=0)
  g1[5] = (int)row_stride_elems;                         // dim0_stride[31:0]
  g1[6] = 0;                                             // dim0_stride[47:32], dim1_stride lo
  g1[7] = 0;
  const v4i z4 = {0, 0, 0, 0};
#if __clang_major__ >= 23
  const v8i z8 = {0, 0, 0, 0, 0, 0, 0, 0};
  __builtin_amdgcn_tensor_load_to_lds(g0, g1, z4, z4, z8, 0);
#else
  __builtin_amdgcn_tensor_load_to_lds(g0, g1, z4, z4, 0);
#endif
}
#endif

// ---------------------------------------------------------------------------
// Workspace layout (bytes). Total ~109 MB.
// ---------------------------------------------------------------------------
#define WS_WHHT   0u          // [512][2048] bf16  (w_hh transposed, K-major)
#define WS_WIHT   2097152u    // [64][2048]  bf16
#define WS_WSHT   2359296u    // [512][512]  bf16  (w_sh^T)
#define WS_WSAT   2883584u    // [512][512]  bf16
#define WS_WRAT   3407872u    // [512][512]  bf16
#define WS_WSTT   3932160u    // [512][64]   bf16  (w_st^T, col 63 zero-padded)
#define WS_HBUF   3997696u    // 2 x [32][512] bf16 (ping-pong recurrent h)
#define WS_ESA    4063232u    // [32768] f32 attention scores (state)
#define WS_ERA    4194304u    // [32768] f32 attention scores (reward)
#define WS_CTR    4325376u    // unsigned barrier generation counter
#define WS_ROUT   4325632u    // [32768][512] bf16 LSTM outputs
#define WS_SHARED 37880064u   // [32768][512] bf16 shared features (incl. bias)
#define WS_CTX    71434496u   // [32768][512] bf16 state attention context
#define WS_XBF    104988928u  // [32][1024][64] bf16 pre-converted x

// LDS carve offsets inside k_lstm's dynamic shared block (121088 B total)
#define LDS_W_OFF     0u
#define LDS_H_OFF     73728u
#define LDS_X_OFF     106496u
#define LDS_GATES_OFF 110592u
#define LDS_C_OFF     118784u
#define LDS_BIAS_OFF  120832u

// d_out layout (f32): state_outs | reward_outs | hT | cT
#define OUT_ST 0
#define OUT_RW 2064384
#define OUT_HT 2097152
#define OUT_CT 2113536

// ---------------------------------------------------------------------------
// Kernel 0: f32->bf16 weight transposition, x pre-conversion, state reset.
// ---------------------------------------------------------------------------
__global__ void k_convert(const float* __restrict__ w_ih, const float* __restrict__ w_hh,
                          const float* __restrict__ w_sh, const float* __restrict__ w_sa,
                          const float* __restrict__ w_ra, const float* __restrict__ w_st,
                          const float* __restrict__ h0,   const float* __restrict__ bv_sa,
                          const float* __restrict__ bv_ra, const float* __restrict__ x,
                          bf16* whhT, bf16* wihT, bf16* wshT, bf16* wsaT, bf16* wraT,
                          bf16* wstT, bf16* hbuf, float* esa, float* era, bf16* xbf,
                          unsigned* ctr) {
  long i = (long)blockIdx.x * 256 + threadIdx.x;
  const long n_whh = 512L * 2048, n_wih = 64L * 2048, n_sq = 512L * 512;
  const long n_st = 512L * 64, n_h = 32L * 512, n_e = 32768L, n_x = 32L * 1024 * 64;
  if (i < n_whh) { int k = i >> 11, r = i & 2047; whhT[i] = (bf16)w_hh[(long)r * 512 + k]; return; }
  i -= n_whh;
  if (i < n_wih) { int k = i >> 11, r = i & 2047; wihT[i] = (bf16)w_ih[(long)r * 64 + k]; return; }
  i -= n_wih;
  if (i < n_sq)  { int k = i >> 9, n = i & 511; wshT[i] = (bf16)w_sh[(long)n * 512 + k]; return; }
  i -= n_sq;
  if (i < n_sq)  { int k = i >> 9, n = i & 511; wsaT[i] = (bf16)w_sa[(long)n * 512 + k]; return; }
  i -= n_sq;
  if (i < n_sq)  { int k = i >> 9, n = i & 511; wraT[i] = (bf16)w_ra[(long)n * 512 + k]; return; }
  i -= n_sq;
  if (i < n_st)  { int k = i >> 6, n = i & 63; wstT[i] = (n < F_) ? (bf16)w_st[(long)n * 512 + k] : (bf16)0.f; return; }
  i -= n_st;
  if (i < n_h)   { hbuf[i] = (bf16)h0[i]; return; }   // ping buffer 0 = h0
  i -= n_h;
  if (i < n_e)   { esa[i] = bv_sa[0]; return; }
  i -= n_e;
  if (i < n_e)   { era[i] = bv_ra[0]; return; }
  i -= n_e;
  if (i < n_x)   { xbf[i] = (bf16)x[i]; return; }
  i -= n_x;
  if (i == 0)    { *ctr = 0u; }
}

// ---------------------------------------------------------------------------
// Kernel 1: persistent WMMA LSTM.
// 32 workgroups x 256 threads (8 waves). WG bw owns h-columns [bw*16, bw*16+16),
// i.e. 64 gate rows {i,f,g,o}. Weight slice [576 K][64 N] bf16 stays in LDS.
// Per step: TDM (TENSOR_LOAD_TO_LDS) stages h (ping-pong global bf16) and x_t
// into LDS, 18 WMMAs per wave (16 K-steps over h, 2 over x), gate epilogue,
// then a device-scope generation barrier. Dynamic LDS = 121088 B (< 320 KB).
// ---------------------------------------------------------------------------
__global__ void __launch_bounds__(256) k_lstm(
    const bf16* __restrict__ xbf, const float* __restrict__ c0,
    const float* __restrict__ b_ih, const float* __restrict__ b_hh,
    const bf16* __restrict__ whhT, const bf16* __restrict__ wihT,
    bf16* __restrict__ hbuf, bf16* __restrict__ rout,
    float* __restrict__ outHT, float* __restrict__ outCT,
    unsigned* __restrict__ ctr) {
  extern __shared__ char smem[];
  bf16*  lds_w     = (bf16*)(smem + LDS_W_OFF);      // [576][64]  73728 B
  bf16*  lds_h     = (bf16*)(smem + LDS_H_OFF);      // [32][512]  32768 B
  bf16*  lds_x     = (bf16*)(smem + LDS_X_OFF);      // [32][64]    4096 B
  float* lds_gates = (float*)(smem + LDS_GATES_OFF); // [32][64]    8192 B
  float* lds_c     = (float*)(smem + LDS_C_OFF);     // [32][16]    2048 B
  float* lds_bias  = (float*)(smem + LDS_BIAS_OFF);  // [64]         256 B

  const int tid = threadIdx.x;
  const int wid = tid >> 5, lane = tid & 31;
  const int bw  = blockIdx.x;          // 0..31
  const int hcb = bw * 16;             // base h-column

  // Stage resident weight slice: rows 0..511 from w_hh^T, 512..575 from w_ih^T.
  for (int i = tid; i < 576 * 64; i += 256) {
    int k = i >> 6, c = i & 63;
    int r = (c >> 4) * H_ + hcb + (c & 15);   // gate-major global row
    lds_w[i] = (k < 512) ? whhT[(long)k * G4_ + r] : wihT[(long)(k - 512) * G4_ + r];
  }
  if (tid < 64) {
    int r = (tid >> 4) * H_ + hcb + (tid & 15);
    lds_bias[tid] = b_ih[r] + b_hh[r];
  }
  for (int i = tid; i < 512; i += 256) {
    int b = i >> 4, c = i & 15;
    lds_c[i] = c0[b * H_ + hcb + c];
  }
  __syncthreads();

  const int m0 = (wid & 1) * 16;       // 2 M-tiles (B=32)
  const int n0 = (wid >> 1) * 16;      // 4 N-tiles (64 gate cols)

  for (int t = 0; t < T_; ++t) {
    const int cur = t & 1, nxt = cur ^ 1;
#if HAVE_TDM
    // Tensor Data Mover: stage h (32x512 tile) and x_t (32x64 strided tile).
    if (wid == 0) {
      tdm_load_2d_bf16(LDS_H_OFF, hbuf + cur * (B_ * H_), H_, B_, H_);
      tdm_load_2d_bf16(LDS_X_OFF, xbf + (long)t * I_, I_, B_, (unsigned)(T_ * I_));
    }
    __builtin_amdgcn_s_wait_tensorcnt(0);
#else
    {
      const v8bf* src = (const v8bf*)(hbuf + cur * (B_ * H_));
      v8bf* dst = (v8bf*)lds_h;
      for (int i = tid; i < (B_ * H_) / 8; i += 256) dst[i] = src[i];
      const v8bf* xs = (const v8bf*)(xbf + (long)0);
      for (int i = tid; i < (B_ * I_) / 8; i += 256) {
        int b = i >> 3, blk = i & 7;   // 8 v8bf chunks per row of 64
        ((v8bf*)lds_x)[i] = *(const v8bf*)(xbf + ((long)b * T_ + t) * I_ + blk * 8);
      }
      (void)xs;
    }
#endif
    __syncthreads();

    v8f acc = {};
#pragma unroll
    for (int ks = 0; ks < 16; ++ks) {                 // recurrent part, K=512
      v16bf a = load_a_tile(lds_h, H_, m0, ks * 32, lane);
      v16bf b = load_b_tile(lds_w, 64, ks * 32, n0, lane);
      acc = wmma_bf16(a, b, acc);
    }
#pragma unroll
    for (int ks = 0; ks < 2; ++ks) {                  // input part, K=64
      v16bf a = load_a_tile(lds_x, I_, m0, ks * 32, lane);
      v16bf b = load_b_tile(lds_w, 64, 512 + ks * 32, n0, lane);
      acc = wmma_bf16(a, b, acc);
    }
#pragma unroll
    for (int v = 0; v < 8; ++v)
      lds_gates[(m0 + v + ((lane < 16) ? 0 : 8)) * 64 + n0 + (lane & 15)] = acc[v];
    __syncthreads();

    // Gate epilogue: 512 (b, col) elements, 2 per thread.
#pragma unroll
    for (int e = 0; e < 2; ++e) {
      int idx = tid + e * 256;
      int b = idx >> 4, c = idx & 15;
      float gi = lds_gates[b * 64 + c]      + lds_bias[c];
      float gf = lds_gates[b * 64 + 16 + c] + lds_bias[16 + c];
      float gg = lds_gates[b * 64 + 32 + c] + lds_bias[32 + c];
      float go = lds_gates[b * 64 + 48 + c] + lds_bias[48 + c];
      float si = 1.f / (1.f + __expf(-gi));
      float sf = 1.f / (1.f + __expf(-gf));
      float sg = tanhf(gg);
      float so = 1.f / (1.f + __expf(-go));
      float cn = sf * lds_c[idx] + si * sg;
      lds_c[idx] = cn;
      float hn = so * tanhf(cn);
      int hc = hcb + c;
      hbuf[nxt * (B_ * H_) + b * H_ + hc] = (bf16)hn;
      rout[((long)b * T_ + t) * H_ + hc]  = (bf16)hn;
      if (t == T_ - 1) { outHT[b * H_ + hc] = hn; outCT[b * H_ + hc] = cn; }
    }
    __threadfence();
    __syncthreads();
    // Device-scope generation barrier across the 32 persistent workgroups.
    if (tid == 0) {
      __hip_atomic_fetch_add(ctr, 1u, __ATOMIC_ACQ_REL, __HIP_MEMORY_SCOPE_AGENT);
      const unsigned tgt = 32u * (unsigned)(t + 1);
      while (__hip_atomic_load(ctr, __ATOMIC_ACQUIRE, __HIP_MEMORY_SCOPE_AGENT) < tgt)
        __builtin_amdgcn_s_sleep(2);
    }
    __syncthreads();
  }
}

// ---------------------------------------------------------------------------
// Kernel 2: shared = r_out @ w_sh^T + b_sh  (32768x512x512), bf16 out.
// One 16x16 C tile per wave, K=512 -> 16 WMMAs.
// ---------------------------------------------------------------------------
__global__ void __launch_bounds__(256) k_shared(const bf16* __restrict__ A,
                                                const bf16* __restrict__ Bw,
                                                const float* __restrict__ bias,
                                                bf16* __restrict__ out) {
  const int wid = threadIdx.x >> 5, lane = threadIdx.x & 31;
  const int tile = blockIdx.x * 8 + wid;
  const int m0 = (tile >> 5) << 4;   // 32 N-tiles
  const int n0 = (tile & 31) << 4;
  v8f acc = {};
#pragma unroll
  for (int ks = 0; ks < 16; ++ks) {
    v16bf a = load_a_tile(A,  H_, m0, ks * 32, lane);
    v16bf b = load_b_tile(Bw, H_, ks * 32, n0, lane);
    acc = wmma_bf16(a, b, acc);
  }
  const int col = n0 + (lane & 15);
  const float bb = bias[col];
  const int rb = m0 + ((lane < 16) ? 0 : 8);
#pragma unroll
  for (int v = 0; v < 8; ++v)
    out[(long)(rb + v) * H_ + col] = (bf16)(acc[v] + bb);
}

// ---------------------------------------------------------------------------
// Kernel 3: attention scores e = tanh(shared @ W^T + b) @ v^T + bv.
// Fused epilogue: tanh, scale by v[col], cross-lane reduce over the 16 columns,
// atomicAdd per-row partials into e (pre-initialized to bv by k_convert).
// ---------------------------------------------------------------------------
__global__ void __launch_bounds__(256) k_escore(const bf16* __restrict__ A,
                                                const bf16* __restrict__ Bw,
                                                const float* __restrict__ bias,
                                                const float* __restrict__ vvec,
                                                float* __restrict__ e) {
  const int wid = threadIdx.x >> 5, lane = threadIdx.x & 31;
  const int tile = blockIdx.x * 8 + wid;
  const int m0 = (tile >> 5) << 4;
  const int n0 = (tile & 31) << 4;
  v8f acc = {};
#pragma unroll
  for (int ks = 0; ks < 16; ++ks) {
    v16bf a = load_a_tile(A,  H_, m0, ks * 32, lane);
    v16bf b = load_b_tile(Bw, H_, ks * 32, n0, lane);
    acc = wmma_bf16(a, b, acc);
  }
  const int col = n0 + (lane & 15);
  const float bb = bias[col], vv = vvec[col];
  float p[8];
#pragma unroll
  for (int v = 0; v < 8; ++v) p[v] = tanhf(acc[v] + bb) * vv;
#pragma unroll
  for (int v = 0; v < 8; ++v) {
    p[v] += __shfl_xor(p[v], 1);
    p[v] += __shfl_xor(p[v], 2);
    p[v] += __shfl_xor(p[v], 4);
    p[v] += __shfl_xor(p[v], 8);
  }
  if ((lane & 15) == 0) {
    const int rb = m0 + ((lane < 16) ? 0 : 8);
#pragma unroll
    for (int v = 0; v < 8; ++v) atomicAdd(&e[rb + v], p[v]);
  }
}

// ---------------------------------------------------------------------------
// Kernel 4: linear-time causal prefix-softmax pooling (flash-style scan).
// ctx_t = V_t / S_t with running (max M, sum S, accum V). 64 blocks:
// blockIdx = b + 32*kind (kind 0 = state ctx -> CTX bf16; 1 = reward -> d_out).
// ---------------------------------------------------------------------------
__global__ void __launch_bounds__(512) k_scan(const bf16* __restrict__ shared_f,
                                              const float* __restrict__ esa,
                                              const float* __restrict__ era,
                                              const float* __restrict__ wrw,
                                              const float* __restrict__ brw,
                                              bf16* __restrict__ ctx_out,
                                              float* __restrict__ rew_out) {
  __shared__ float red[16];
  const int b = blockIdx.x & 31;
  const int kind = blockIdx.x >> 5;
  const int d = threadIdx.x;           // one h-dim per thread
  const float* e = kind ? era : esa;
  const float wr = wrw[d];
  float M = 0.f, S = 0.f, V = 0.f;
  for (int t = 0; t < T_; ++t) {
    const float s  = (float)shared_f[((long)b * T_ + t) * H_ + d];
    const float ec = e[b * T_ + t];
    if (t == 0)        { M = ec; S = 1.f; V = s; }
    else if (ec > M)   { float sc = __expf(M - ec); S = S * sc + 1.f; V = V * sc + s; M = ec; }
    else               { float p = __expf(ec - M);  S += p;           V += p * s; }
    const float ctx = V / S;
    if (kind == 0) {
      ctx_out[((long)b * T_ + t) * H_ + d] = (bf16)ctx;
    } else {
      // reward_t = dot(ctx, w_rw) + b_rw, reduced across the 512 threads.
      float prod = ctx * wr;
      prod += __shfl_xor(prod, 16);
      prod += __shfl_xor(prod, 8);
      prod += __shfl_xor(prod, 4);
      prod += __shfl_xor(prod, 2);
      prod += __shfl_xor(prod, 1);
      if ((threadIdx.x & 31) == 0) red[threadIdx.x >> 5] = prod;
      __syncthreads();
      if (threadIdx.x == 0) {
        float sum = 0.f;
#pragma unroll
        for (int i = 0; i < 16; ++i) sum += red[i];
        rew_out[b * T_ + t] = sum + brw[0];
      }
      __syncthreads();
    }
  }
}

// ---------------------------------------------------------------------------
// Kernel 5: state head  state_outs = ctx @ w_st^T + b_st  (N=63 padded to 64).
// ---------------------------------------------------------------------------
__global__ void __launch_bounds__(256) k_head(const bf16* __restrict__ A,
                                              const bf16* __restrict__ Bw,
                                              const float* __restrict__ bias,
                                              float* __restrict__ out) {
  const int wid = threadIdx.x >> 5, lane = threadIdx.x & 31;
  const int tile = blockIdx.x * 8 + wid;
  const int m0 = (tile >> 2) << 4;   // 4 N-tiles
  const int n0 = (tile & 3) << 4;
  v8f acc = {};
#pragma unroll
  for (int ks = 0; ks < 16; ++ks) {
    v16bf a = load_a_tile(A,  H_, m0, ks * 32, lane);
    v16bf b = load_b_tile(Bw, 64, ks * 32, n0, lane);
    acc = wmma_bf16(a, b, acc);
  }
  const int col = n0 + (lane & 15);
  if (col < F_) {
    const float bb = bias[col];
    const int rb = m0 + ((lane < 16) ? 0 : 8);
#pragma unroll
    for (int v = 0; v < 8; ++v)
      out[(long)(rb + v) * F_ + col] = acc[v] + bb;
  }
}

// ---------------------------------------------------------------------------
// Host launcher
// ---------------------------------------------------------------------------
extern "C" void kernel_launch(void* const* d_in, const int* in_sizes, int n_in,
                              void* d_out, int out_size, void* d_ws, size_t ws_size,
                              hipStream_t stream) {
  (void)in_sizes; (void)n_in; (void)out_size; (void)ws_size;
  const float* x     = (const float*)d_in[0];
  // d_in[1] = x_lengths (all T -> pack/pad is identity, unused)
  const float* h0    = (const float*)d_in[2];
  const float* c0    = (const float*)d_in[3];
  const float* w_ih  = (const float*)d_in[4];
  const float* w_hh  = (const float*)d_in[5];
  const float* b_ih  = (const float*)d_in[6];
  const float* b_hh  = (const float*)d_in[7];
  const float* w_sh  = (const float*)d_in[8];
  const float* b_sh  = (const float*)d_in[9];
  const float* w_sa  = (const float*)d_in[10];
  const float* b_sa  = (const float*)d_in[11];
  const float* v_sa  = (const float*)d_in[12];
  const float* bv_sa = (const float*)d_in[13];
  const float* w_ra  = (const float*)d_in[14];
  const float* b_ra  = (const float*)d_in[15];
  const float* v_ra  = (const float*)d_in[16];
  const float* bv_ra = (const float*)d_in[17];
  const float* w_st  = (const float*)d_in[18];
  const float* b_st  = (const float*)d_in[19];
  const float* w_rw  = (const float*)d_in[20];
  const float* b_rw  = (const float*)d_in[21];

  char* ws = (char*)d_ws;
  bf16*     WHHT   = (bf16*)(ws + WS_WHHT);
  bf16*     WIHT   = (bf16*)(ws + WS_WIHT);
  bf16*     WSHT   = (bf16*)(ws + WS_WSHT);
  bf16*     WSAT   = (bf16*)(ws + WS_WSAT);
  bf16*     WRAT   = (bf16*)(ws + WS_WRAT);
  bf16*     WSTT   = (bf16*)(ws + WS_WSTT);
  bf16*     HBUF   = (bf16*)(ws + WS_HBUF);
  float*    ESA    = (float*)(ws + WS_ESA);
  float*    ERA    = (float*)(ws + WS_ERA);
  unsigned* CTR    = (unsigned*)(ws + WS_CTR);
  bf16*     ROUT   = (bf16*)(ws + WS_ROUT);
  bf16*     SHARED = (bf16*)(ws + WS_SHARED);
  bf16*     CTX    = (bf16*)(ws + WS_CTX);
  bf16*     XBF    = (bf16*)(ws + WS_XBF);

  float* out = (float*)d_out;

  // 0) weight conversion + x bf16 conversion + per-call state reset
  const long total0 = 512L * 2048 + 64L * 2048 + 3L * 512 * 512 + 512L * 64
                    + 32L * 512 + 2L * 32768 + 32L * 1024 * 64 + 1;
  k_convert<<<(int)((total0 + 255) / 256), 256, 0, stream>>>(
      w_ih, w_hh, w_sh, w_sa, w_ra, w_st, h0, bv_sa, bv_ra, x,
      WHHT, WIHT, WSHT, WSAT, WRAT, WSTT, HBUF, ESA, ERA, XBF, CTR);

  // 1) persistent WMMA LSTM with TDM staging (32 WGs, 121088 B dynamic LDS)
  k_lstm<<<32, 256, 121088, stream>>>(XBF, c0, b_ih, b_hh, WHHT, WIHT, HBUF, ROUT,
                                      out + OUT_HT, out + OUT_CT, CTR);

  // 2) shared projection GEMM (65536 wave-tiles / 8 per block)
  k_shared<<<8192, 256, 0, stream>>>(ROUT, WSHT, b_sh, SHARED);

  // 3) attention score GEMMs with fused tanh * v reduction
  k_escore<<<8192, 256, 0, stream>>>(SHARED, WSAT, b_sa, v_sa, ESA);
  k_escore<<<8192, 256, 0, stream>>>(SHARED, WRAT, b_ra, v_ra, ERA);

  // 4) O(B*T*H) causal prefix-softmax scan (state ctx + fused reward head)
  k_scan<<<64, 512, 0, stream>>>(SHARED, ESA, ERA, w_rw, b_rw, CTX, out + OUT_RW);

  // 5) state head GEMM (F=63 padded to 64)
  k_head<<<1024, 256, 0, stream>>>(CTX, WSTT, b_st, out + OUT_ST);
}
